// CausalSelfAttention_19284403159376
// MI455X (gfx1250) — matrix operand order repaired and checked by reading, not compiled, not run on hardware
//
#include <hip/hip_runtime.h>
#include <hip/hip_bf16.h>
#include <stdint.h>

// ---- problem constants ----
#define NB 2
#define TT 2048
#define CC 1024
#define HH 16
#define DD 64
#define MM (NB * TT)          // 4096 rows of x

typedef _Float16 h8  __attribute__((ext_vector_type(8)));
typedef _Float16 h16 __attribute__((ext_vector_type(16)));
typedef float    f8  __attribute__((ext_vector_type(8)));
typedef int      v4i __attribute__((ext_vector_type(4)));

__device__ __forceinline__ h16 cat8(h8 a, h8 b) {
  return __builtin_shufflevector(a, b, 0,1,2,3,4,5,6,7,8,9,10,11,12,13,14,15);
}

__device__ __forceinline__ f8 wmma_f16(h16 a, h16 b, f8 c) {
  return __builtin_amdgcn_wmma_f32_16x16x32_f16(false, a, false, b, (short)0, c, false, false);
}

// ---- async global -> LDS staging (guarded; falls back to plain LDS stores) ----
#if defined(__has_builtin)
#if __has_builtin(__builtin_amdgcn_global_load_async_to_lds_b128)
#define HAVE_ASYNC_LDS 1
#endif
#endif

#define AS1 __attribute__((address_space(1)))
#define AS3 __attribute__((address_space(3)))

__device__ __forceinline__ AS1 v4i* gbl_cast(const void* p) {
  return (AS1 v4i*)(uintptr_t)p;
}
__device__ __forceinline__ AS3 v4i* lds_cast(void* p) {
  // generic LDS addr: low 32 bits are the LDS offset (ISA aperture rule)
  return (AS3 v4i*)(uint32_t)(uintptr_t)p;
}

__device__ __forceinline__ void wait_async0() {
#if defined(__has_builtin)
#if __has_builtin(__builtin_amdgcn_s_wait_asynccnt)
  __builtin_amdgcn_s_wait_asynccnt(0);
  return;
#endif
#endif
  asm volatile("s_wait_asynccnt 0" ::: "memory");
}

// ---------------- f32 -> f16 convert ----------------
__global__ __launch_bounds__(256)
void cvt_f16(const float* __restrict__ src, _Float16* __restrict__ dst, int n) {
  int i = blockIdx.x * 256 + threadIdx.x;
  if (i < n) dst[i] = (_Float16)src[i];
}

// ---------------- GEMM core ----------------
// out[m,n] = sum_k A[m,k] * W[n,k]; A panel (32 rows) staged in LDS per 512-k
// phase via async copy; each of 8 waves computes 32m x 64n (8 WMMA / k-step).
#define KPHASE 512
#define APAD   8
#define ASTRIDE (KPHASE + APAD)    // 520 halves: 1040B ~ 4-bank skew, conflict-free

#define GEMM_MAINLOOP(AH)                                                        \
  __shared__ __align__(16) _Float16 atile[32][ASTRIDE];                          \
  const int lane = threadIdx.x & 31;                                             \
  const int wave = threadIdx.x >> 5;                                             \
  const int hg   = lane >> 4;                                                    \
  const int col  = lane & 15;                                                    \
  const int m0   = blockIdx.x * 32;                                              \
  const int n0   = blockIdx.y * 512 + wave * 64;                                 \
  f8 acc[2][4] = {};                                                             \
  for (int kp = 0; kp < CC; kp += KPHASE) {                                      \
    if (kp) __syncthreads();                                                     \
    _Pragma("unroll")                                                            \
    for (int i = 0; i < 8; ++i) {                                                \
      int c = threadIdx.x + 256 * i;                                             \
      int row = c >> 6;                                                          \
      int coh = (c & 63) * 8;                                                    \
      const _Float16* g = (AH) + (size_t)(m0 + row) * CC + kp + coh;             \
      _Float16* l = &atile[row][coh];                                            \
      STAGE16(g, l);                                                             \
    }                                                                            \
    STAGE_FENCE();                                                               \
    __syncthreads();                                                             \
    for (int k0 = 0; k0 < KPHASE; k0 += 32) {                                    \
      h16 a0 = cat8(*(const h8*)(&atile[col][k0 + hg * 16]),                     \
                    *(const h8*)(&atile[col][k0 + hg * 16 + 8]));                \
      h16 a1 = cat8(*(const h8*)(&atile[col + 16][k0 + hg * 16]),                \
                    *(const h8*)(&atile[col + 16][k0 + hg * 16 + 8]));           \
      _Pragma("unroll")                                                          \
      for (int j = 0; j < 4; ++j) {                                              \
        int n = n0 + j * 16 + col;                                               \
        h16 b = *(const h16*)(wh + (size_t)n * CC + kp + k0 + hg * 16);          \
        acc[0][j] = wmma_f16(a0, b, acc[0][j]);                                  \
        acc[1][j] = wmma_f16(a1, b, acc[1][j]);                                  \
      }                                                                          \
    }                                                                            \
  }

#ifdef HAVE_ASYNC_LDS
#define STAGE16(g, l) \
  __builtin_amdgcn_global_load_async_to_lds_b128(gbl_cast(g), lds_cast(l), 0, 0)
#define STAGE_FENCE() wait_async0()
#else
#define STAGE16(g, l) (*(h8*)(l) = *(const h8*)(g))
#define STAGE_FENCE() ((void)0)
#endif

// ---------------- QKV projection GEMM ----------------
// M=4096, K=1024, N=3072. Q,K stored [B,H,T,D] f16; V transposed [B,H,D,T] f16.
__global__ __launch_bounds__(256)
void qkv_gemm(const _Float16* __restrict__ xh, const _Float16* __restrict__ wh,
              const float* __restrict__ bias,
              _Float16* __restrict__ Qh, _Float16* __restrict__ Kh,
              _Float16* __restrict__ Vth) {
  GEMM_MAINLOOP(xh)
#pragma unroll
  for (int j = 0; j < 4; ++j) {
    int n = n0 + j * 16 + col;
    float bv = bias[n];
    int region = n >> 10;           // 0=Q 1=K 2=V
    int c = n & 1023;
    int h = c >> 6, d = c & 63;
#pragma unroll
    for (int g = 0; g < 2; ++g)
#pragma unroll
      for (int r = 0; r < 8; ++r) {
        int tm = m0 + g * 16 + r + 8 * hg;
        int bb = tm >> 11, t = tm & 2047;
        int bh = bb * HH + h;
        _Float16 v = (_Float16)(acc[g][j][r] + bv);
        if (region == 0)      Qh[((size_t)bh * TT + t) * DD + d] = v;
        else if (region == 1) Kh[((size_t)bh * TT + t) * DD + d] = v;
        else                  Vth[((size_t)bh * DD + d) * TT + t] = v;
      }
  }
}

// ---------------- output projection GEMM ----------------
__global__ __launch_bounds__(256)
void proj_gemm(const _Float16* __restrict__ ah, const _Float16* __restrict__ wh,
               const float* __restrict__ bias, float* __restrict__ out) {
  GEMM_MAINLOOP(ah)
#pragma unroll
  for (int j = 0; j < 4; ++j) {
    int n = n0 + j * 16 + col;
    float bv = bias[n];
#pragma unroll
    for (int g = 0; g < 2; ++g)
#pragma unroll
      for (int r = 0; r < 8; ++r) {
        int tm = m0 + g * 16 + r + 8 * hg;
        out[(size_t)tm * CC + n] = acc[g][j][r] + bv;
      }
  }
}

// ---------------- causal flash attention ----------------
// One wave per 16-query tile; 4 waves / block. Online softmax, 32 keys/iter.
__global__ __launch_bounds__(128)
void flash_attn(const _Float16* __restrict__ Qh, const _Float16* __restrict__ Kh,
                const _Float16* __restrict__ Vth, _Float16* __restrict__ attnh) {
  __shared__ __align__(64) _Float16 plds[4][16][32];
  const int lane = threadIdx.x & 31;
  const int wave = threadIdx.x >> 5;
  const int hg   = lane >> 4;
  const int col  = lane & 15;

  const int tile   = blockIdx.x * 4 + wave;
  const int qtiles = TT / 16;
  const int bh = tile / qtiles;
  const int qb = (tile % qtiles) * 16;

  const _Float16* qbase = Qh + ((size_t)bh * TT + qb + col) * DD;
  const h16 qf0 = cat8(*(const h8*)(qbase + hg * 16),      *(const h8*)(qbase + 8 + hg * 16));
  const h16 qf1 = cat8(*(const h8*)(qbase + 32 + hg * 16), *(const h8*)(qbase + 40 + hg * 16));

  const _Float16* kbase = Kh  + (size_t)bh * TT * DD;
  const _Float16* vbase = Vth + (size_t)bh * DD * TT;

  f8 o[4] = {};
  f8 mi, li = {};
#pragma unroll
  for (int r = 0; r < 8; ++r) mi[r] = -__builtin_inff();

  const float sc = 0.125f * 1.44269504088896340736f;   // log2(e)/sqrt(D)
  const int kend = qb + 16;

  for (int kb = 0; kb < kend; kb += 32) {
    // prefetch next key/value tiles into cache while we do softmax VALU work
    if (kb + 32 < kend) {
      __builtin_prefetch((const void*)(kbase + (size_t)(kb + 32 + col) * DD), 0, 3);
      __builtin_prefetch((const void*)(vbase + (size_t)col * TT + kb + 32), 0, 3);
    }
    // ---- S = Q K^T (two 16-key column tiles) ----
    f8 s0 = {}, s1 = {};
    {
      const _Float16* kr0 = kbase + (size_t)(kb + col) * DD;
      const _Float16* kr1 = kbase + (size_t)(kb + 16 + col) * DD;
      s0 = wmma_f16(qf0, *(const h16*)(kr0 + hg * 16), s0);
      s0 = wmma_f16(qf1, *(const h16*)(kr0 + 32 + hg * 16), s0);
      s1 = wmma_f16(qf0, *(const h16*)(kr1 + hg * 16), s1);
      s1 = wmma_f16(qf1, *(const h16*)(kr1 + 32 + hg * 16), s1);
    }
    // ---- scale + causal mask ----
#pragma unroll
    for (int r = 0; r < 8; ++r) {
      int tq = qb + r + 8 * hg;
      s0[r] = (kb + col      <= tq) ? s0[r] * sc : -__builtin_inff();
      s1[r] = (kb + 16 + col <= tq) ? s1[r] * sc : -__builtin_inff();
    }
    // ---- online softmax: row max over 16 lanes of half-group ----
    f8 mn;
#pragma unroll
    for (int r = 0; r < 8; ++r) mn[r] = fmaxf(mi[r], fmaxf(s0[r], s1[r]));
#pragma unroll
    for (int off = 1; off < 16; off <<= 1)
#pragma unroll
      for (int r = 0; r < 8; ++r) mn[r] = fmaxf(mn[r], __shfl_xor(mn[r], off, 32));

    f8 alpha, p0, p1, rs;
#pragma unroll
    for (int r = 0; r < 8; ++r) {
      alpha[r] = exp2f(mi[r] - mn[r]);
      p0[r]    = exp2f(s0[r] - mn[r]);
      p1[r]    = exp2f(s1[r] - mn[r]);
      rs[r]    = p0[r] + p1[r];
      mi[r]    = mn[r];
    }
#pragma unroll
    for (int off = 1; off < 16; off <<= 1)
#pragma unroll
      for (int r = 0; r < 8; ++r) rs[r] += __shfl_xor(rs[r], off, 32);
#pragma unroll
    for (int r = 0; r < 8; ++r) {
      li[r] = li[r] * alpha[r] + rs[r];
      o[0][r] *= alpha[r]; o[1][r] *= alpha[r];
      o[2][r] *= alpha[r]; o[3][r] *= alpha[r];
    }
    // ---- reshape P (C layout) -> A fragment via per-wave LDS slot ----
#pragma unroll
    for (int r = 0; r < 8; ++r) {
      plds[wave][r + 8 * hg][col]      = (_Float16)p0[r];
      plds[wave][r + 8 * hg][col + 16] = (_Float16)p1[r];
    }
    asm volatile("s_wait_dscnt 0" ::: "memory");
    const _Float16* prow = &plds[wave][col][0];
    h16 pf = cat8(*(const h8*)(prow + hg * 16), *(const h8*)(prow + 8 + hg * 16));
    // ---- O += P V  (V transposed: contiguous key-run per lane) ----
    const _Float16* vk = vbase + kb + hg * 16;
#pragma unroll
    for (int j = 0; j < 4; ++j) {
      int d = j * 16 + col;
      h16 vf = *(const h16*)(vk + (size_t)d * TT);
      o[j] = wmma_f16(pf, vf, o[j]);
    }
  }

  // ---- normalize and write attn output as f16 [B,T,C] ----
  f8 inv;
#pragma unroll
  for (int r = 0; r < 8; ++r) inv[r] = 1.0f / li[r];
  const int b = bh >> 4, h = bh & 15;
#pragma unroll
  for (int j = 0; j < 4; ++j) {
    int cidx = h * DD + j * 16 + col;
#pragma unroll
    for (int r = 0; r < 8; ++r) {
      int t = qb + r + 8 * hg;
      attnh[((size_t)b * TT + t) * CC + cidx] = (_Float16)(o[j][r] * inv[r]);
    }
  }
}

// ---------------- launcher ----------------
extern "C" void kernel_launch(void* const* d_in, const int* in_sizes, int n_in,
                              void* d_out, int out_size, void* d_ws, size_t ws_size,
                              hipStream_t stream) {
  (void)in_sizes; (void)n_in; (void)out_size; (void)ws_size;
  const float* x      = (const float*)d_in[0];
  const float* w_qkv  = (const float*)d_in[1];
  const float* b_qkv  = (const float*)d_in[2];
  const float* w_proj = (const float*)d_in[3];
  const float* b_proj = (const float*)d_in[4];
  float* out = (float*)d_out;

  // workspace layout (halves)
  _Float16* xh     = (_Float16*)d_ws;                 // M*C      = 4M
  _Float16* wqkvh  = xh     + (size_t)MM * CC;        // 3C*C     = 3M
  _Float16* wprojh = wqkvh  + (size_t)3 * CC * CC;    // C*C      = 1M
  _Float16* Qh     = wprojh + (size_t)CC * CC;        // B*H*T*D  = 4M
  _Float16* Kh     = Qh     + (size_t)NB * HH * TT * DD;
  _Float16* Vth    = Kh     + (size_t)NB * HH * TT * DD;
  _Float16* attnh  = Vth    + (size_t)NB * HH * TT * DD;  // B*T*C = 4M

  const int nX = MM * CC, nWq = 3 * CC * CC, nWp = CC * CC;
  cvt_f16<<<(nX  + 255) / 256, 256, 0, stream>>>(x,      xh,     nX);
  cvt_f16<<<(nWq + 255) / 256, 256, 0, stream>>>(w_qkv,  wqkvh,  nWq);
  cvt_f16<<<(nWp + 255) / 256, 256, 0, stream>>>(w_proj, wprojh, nWp);

  qkv_gemm<<<dim3(MM / 32, 6), 256, 0, stream>>>(xh, wqkvh, b_qkv, Qh, Kh, Vth);
  flash_attn<<<(NB * HH * (TT / 16)) / 4, 128, 0, stream>>>(Qh, Kh, Vth, attnh);
  proj_gemm<<<dim3(MM / 32, 2), 256, 0, stream>>>(attnh, wprojh, b_proj, out);
}
